// SetAbstraction_438086664390
// MI455X (gfx1250) — matrix operand order repaired and checked by reading, not compiled, not run on hardware
//
#include <hip/hip_runtime.h>

// ---------------------------------------------------------------------------
// PointNet++ SetAbstraction for MI455X (gfx1250, wave32, WMMA)
// FPS -> ball-query/group -> 3x (WMMA GEMM + BN stats) -> BN/ReLU -> maxpool
// All GEMM operands live in pre-swizzled WMMA fragment layouts so every
// A/B fragment is one aligned 32-byte load.
// ---------------------------------------------------------------------------

typedef _Float16 f16;
typedef __attribute__((ext_vector_type(16))) _Float16 v16h;
typedef __attribute__((ext_vector_type(8)))  float    v8f;

#define BATCH    8
#define NPTS     8192
#define NPOINT   2048
#define NSAMPLE  32
#define CFEAT    64
#define R2       0.04f
#define EPSBN    1e-5f
#define NCOLS    ((size_t)BATCH * NPOINT * NSAMPLE)   // 524288 GEMM columns
#define NTILES   ((size_t)(NCOLS / 16))               // 32768 column tiles
#define KPAD0    96                                   // 67 -> 3 K-chunks of 32

// A-fragment K mapping (matches B-fragment K order):
//   lanes 0-15 : h=0..15 -> K=0..15
//   lanes 16-31: h=0..15 -> K=16..31
__device__ __host__ inline int ka_of(int lane, int h) {
    return h + ((lane >> 4) << 4);
}
// B-fragment layout (contiguous K order used here: lanes 0-15 hold K=0..15, lanes 16-31
// hold K=16..31; h sequential). Element (k, col) -> half offset inside the
// [kc][n_tile][lane][16] buffer:
__device__ inline size_t bfrag_addr(int k, size_t col) {
    const int    kc   = k >> 5;
    const int    kp   = k & 31;           // lgrp = kp>>4, h = kp&15
    const size_t nt   = col >> 4;
    const int    l16  = (int)(col & 15);
    return (((size_t)kc * NTILES + nt) * 32 + ((kp >> 4) * 16 + l16)) * 16 + (kp & 15);
}

// 64-bit shfl_xor from two 32-bit shuffles (wave32)
__device__ inline unsigned long long shflxor_u64(unsigned long long v, int m) {
    unsigned lo = (unsigned)v, hi = (unsigned)(v >> 32);
    lo = __shfl_xor(lo, m, 32);
    hi = __shfl_xor(hi, m, 32);
    return ((unsigned long long)hi << 32) | lo;
}

// ---------------------------------------------------------------------------
// Kernel 0: zero the BN-stat accumulators (512 floats)
// ---------------------------------------------------------------------------
__global__ void zero_stats_kernel(float* stats) { stats[threadIdx.x] = 0.0f; }

// ---------------------------------------------------------------------------
// Kernel 1: pack W (f32, M x Kreal) into A-fragment f16 layout
//           [m_tile][kc][lane][16], zero-padded where k >= Kreal.
// ---------------------------------------------------------------------------
__global__ __launch_bounds__(256)
void pack_w_kernel(const float* __restrict__ W, f16* __restrict__ Apack,
                   int Mtiles, int KSteps, int Kreal) {
    int e = blockIdx.x * blockDim.x + threadIdx.x;
    if (e >= Mtiles * KSteps * 512) return;
    int h    = e & 15;
    int lane = (e >> 4) & 31;
    int frag = e >> 9;                    // m_tile*KSteps + kc
    int kc   = frag % KSteps;
    int m    = (frag / KSteps) * 16 + (lane & 15);
    int k    = kc * 32 + ka_of(lane, h);
    Apack[e] = (f16)((k < Kreal) ? W[m * Kreal + k] : 0.0f);
}

// ---------------------------------------------------------------------------
// Kernel 2: farthest point sampling (one block per batch, points in regs)
// ---------------------------------------------------------------------------
__global__ __launch_bounds__(1024)
void fps_kernel(const float* __restrict__ xyz, float* __restrict__ new_xyz) {
    const int b = blockIdx.x, tid = threadIdx.x;
    const int lane = tid & 31, wave = tid >> 5;
    __shared__ unsigned long long s_key[32];
    __shared__ int s_far;

    float px[8], py[8], pz[8], dist[8];
    const float* base = xyz + (size_t)b * NPTS * 3;
#pragma unroll
    for (int j = 0; j < 8; ++j) {
        int p = tid + j * 1024;
        px[j] = base[p * 3 + 0]; py[j] = base[p * 3 + 1]; pz[j] = base[p * 3 + 2];
        dist[j] = 1e10f;
    }
    int far = 0;
    for (int it = 0; it < NPOINT; ++it) {
        float cx = base[far * 3 + 0], cy = base[far * 3 + 1], cz = base[far * 3 + 2];
        if (tid == 0) {
            float* o = new_xyz + ((size_t)b * NPOINT + it) * 3;
            o[0] = cx; o[1] = cy; o[2] = cz;
        }
        unsigned long long best = 0ull;
#pragma unroll
        for (int j = 0; j < 8; ++j) {
            int p = tid + j * 1024;
            float dx = px[j] - cx, dy = py[j] - cy, dz = pz[j] - cz;
            dist[j] = fminf(dist[j], dx * dx + dy * dy + dz * dz);
            unsigned long long key =
                ((unsigned long long)__float_as_uint(dist[j]) << 32) |
                (unsigned)(NPTS - 1 - p);          // ties -> smallest index
            best = key > best ? key : best;
        }
#pragma unroll
        for (int o = 16; o >= 1; o >>= 1) {
            unsigned long long v = shflxor_u64(best, o);
            best = v > best ? v : best;
        }
        if (lane == 0) s_key[wave] = best;
        __syncthreads();
        if (wave == 0) {
            unsigned long long v = s_key[lane];
#pragma unroll
            for (int o = 16; o >= 1; o >>= 1) {
                unsigned long long u = shflxor_u64(v, o);
                v = u > v ? u : v;
            }
            if (lane == 0) s_far = NPTS - 1 - (int)(v & 0xffffffffu);
        }
        __syncthreads();
        far = s_far;
    }
}

// ---------------------------------------------------------------------------
// Kernel 3: ball query + gather + concat -> X0 in B-fragment layout.
// One wave per query point; ballot+popc keeps first 32 hits in index order.
// ---------------------------------------------------------------------------
__global__ __launch_bounds__(256)
void group_kernel(const float* __restrict__ xyz, const float* __restrict__ points,
                  const float* __restrict__ new_xyz, f16* __restrict__ X0) {
    __shared__ int sel[8][NSAMPLE];
    const int gw   = (blockIdx.x * blockDim.x + threadIdx.x) >> 5;
    const int w    = (threadIdx.x >> 5) & 7;
    const int lane = threadIdx.x & 31;
    const int b    = gw / NPOINT, s = gw % NPOINT;

    const float* xb = xyz + (size_t)b * NPTS * 3;
    const float* q  = new_xyz + ((size_t)b * NPOINT + s) * 3;
    const float qx = q[0], qy = q[1], qz = q[2];

    int cnt = 0;
    for (int baseP = 0; baseP < NPTS; baseP += 32) {
        int p = baseP + lane;
        float dx = xb[p * 3 + 0] - qx, dy = xb[p * 3 + 1] - qy, dz = xb[p * 3 + 2] - qz;
        bool ok = (dx * dx + dy * dy + dz * dz) <= R2;
        unsigned mask = (unsigned)__ballot(ok);
        int slot = cnt + __popc(mask & ((1u << lane) - 1u));
        if (ok && slot < NSAMPLE) sel[w][slot] = p;
        cnt += __popc(mask);
        if (cnt >= NSAMPLE) break;
    }
    if (cnt == 0) { if (lane == 0) sel[w][0] = 0; cnt = 1; }
    asm volatile("s_wait_dscnt 0" ::: "memory");
    int first = sel[w][0];
    int idx   = (lane < cnt) ? sel[w][lane] : first;

    const size_t col = (size_t)gw * NSAMPLE + lane;
    X0[bfrag_addr(0, col)] = (f16)(xb[idx * 3 + 0] - qx);
    X0[bfrag_addr(1, col)] = (f16)(xb[idx * 3 + 1] - qy);
    X0[bfrag_addr(2, col)] = (f16)(xb[idx * 3 + 2] - qz);
    const float* pb = points + (size_t)b * CFEAT * NPTS;
#pragma unroll 4
    for (int c = 0; c < CFEAT; ++c)
        X0[bfrag_addr(3 + c, col)] = (f16)pb[(size_t)c * NPTS + idx];
#pragma unroll
    for (int c = 3 + CFEAT; c < KPAD0; ++c)
        X0[bfrag_addr(c, col)] = (f16)0.0f;
}

// ---------------------------------------------------------------------------
// Kernel 4: WMMA GEMM. A from packed fragments (regs, reused), 4 n-tiles per
// wave with independent accumulators; D written in B-fragment layout for the
// next layer; per-channel sum/sumsq accumulated for BatchNorm.
// ---------------------------------------------------------------------------
template <int KSteps, int Mtiles>
__global__ __launch_bounds__(256)
void gemm_bnstat_kernel(const f16* __restrict__ Apack, const f16* __restrict__ X,
                        f16* __restrict__ Y, float* __restrict__ sum,
                        float* __restrict__ sumsq) {
    const int wid  = (blockIdx.x * blockDim.x + threadIdx.x) >> 5;
    const int lane = threadIdx.x & 31;
    const int l16  = lane & 15;
    const int hig  = lane >> 4;
    const int m_tile = wid % Mtiles;
    const int nq     = wid / Mtiles;      // quad of column tiles

    v16h a[KSteps];
#pragma unroll
    for (int kc = 0; kc < KSteps; ++kc)
        a[kc] = *(const v16h*)(Apack + (((size_t)(m_tile * KSteps + kc)) * 32 + lane) * 16);

    v8f acc[4];
#pragma unroll
    for (int t = 0; t < 4; ++t) {
        const size_t nt = (size_t)nq * 4 + t;
        v8f c = {};
#pragma unroll
        for (int kc = 0; kc < KSteps; ++kc) {
            v16h bm = *(const v16h*)(X + (((size_t)kc * NTILES + nt) * 32 + lane) * 16);
            c = __builtin_amdgcn_wmma_f32_16x16x32_f16(
                false, a[kc], false, bm, (short)0, c, false, false);
        }
        acc[t] = c;
    }

    // BN stats: per D row, sum value and value^2 over the 64 columns.
    float vs[8], vq[8];
#pragma unroll
    for (int r = 0; r < 8; ++r) {
        float v = acc[0][r] + acc[1][r] + acc[2][r] + acc[3][r];
        float q = acc[0][r] * acc[0][r] + acc[1][r] * acc[1][r] +
                  acc[2][r] * acc[2][r] + acc[3][r] * acc[3][r];
#pragma unroll
        for (int o = 8; o >= 1; o >>= 1) {
            v += __shfl_xor(v, o, 32);
            q += __shfl_xor(q, o, 32);
        }
        vs[r] = v; vq[r] = q;
    }
    if (l16 == 0) {
#pragma unroll
        for (int r = 0; r < 8; ++r) {
            int ch = m_tile * 16 + r + 8 * hig;
            atomicAdd(&sum[ch],   vs[r]);
            atomicAdd(&sumsq[ch], vq[r]);
        }
    }

    // D store, pre-swizzled into next layer's B-fragment layout.
#pragma unroll
    for (int t = 0; t < 4; ++t) {
        const size_t nt = (size_t)nq * 4 + t;
#pragma unroll
        for (int r = 0; r < 8; ++r) {
            int m  = m_tile * 16 + r + 8 * hig;       // output channel
            int kp = m & 31;
            size_t addr = (((size_t)(m >> 5) * NTILES + nt) * 32 +
                           ((kp >> 4) * 16 + l16)) * 16 + (kp & 15);
            Y[addr] = (f16)acc[t][r];
        }
    }
}

// ---------------------------------------------------------------------------
// Kernel 5: per-channel BN scale/shift from accumulated stats
// ---------------------------------------------------------------------------
__global__ void finalize_stats_kernel(const float* __restrict__ sum,
                                      const float* __restrict__ sumsq,
                                      const float* __restrict__ g,
                                      const float* __restrict__ beta,
                                      float* __restrict__ sc, float* __restrict__ sh,
                                      int M) {
    int m = threadIdx.x;
    if (m >= M) return;
    const float inv  = 1.0f / (float)NCOLS;
    const float mean = sum[m] * inv;
    const float var  = sumsq[m] * inv - mean * mean;
    const float s    = g[m] * rsqrtf(var + EPSBN);
    sc[m] = s;
    sh[m] = beta[m] - s * mean;
}

// ---------------------------------------------------------------------------
// Kernel 6: in-place BN + ReLU over a B-fragment buffer.
// One thread per fragment-lane (16 contiguous halves).
// ---------------------------------------------------------------------------
__global__ __launch_bounds__(256)
void bn_relu_kernel(f16* __restrict__ Y, const float* __restrict__ sc,
                    const float* __restrict__ sh) {
    const size_t t    = (size_t)blockIdx.x * blockDim.x + threadIdx.x;
    const size_t tile = t >> 5;
    const int    fl   = (int)(t & 31);
    const int    ch0  = (int)(tile / NTILES) * 32 + ((fl >> 4) << 4); // + h
    f16* p = Y + (t << 4);
#pragma unroll
    for (int h = 0; h < 16; ++h) {
        float v = (float)p[h] * sc[ch0 + h] + sh[ch0 + h];
        p[h] = (f16)fmaxf(v, 0.0f);
    }
}

// ---------------------------------------------------------------------------
// Kernel 7: fused BN + ReLU + max over NSAMPLE, reading B-fragment layout.
// One thread per (b, o, s); its 32 columns span two column tiles.
// ---------------------------------------------------------------------------
__global__ __launch_bounds__(256)
void bn_relu_maxpool_kernel(const f16* __restrict__ Y, const float* __restrict__ sc,
                            const float* __restrict__ sh, float* __restrict__ out) {
    const size_t tid = (size_t)blockIdx.x * blockDim.x + threadIdx.x;
    const int s = (int)(tid % NPOINT);
    const int o = (int)((tid / NPOINT) % 128);
    const int b = (int)(tid / ((size_t)NPOINT * 128));

    const float scale = sc[o], shift = sh[o];
    const int kp = o & 31;
    const size_t col0 = ((size_t)b * NPOINT + s) * NSAMPLE;   // multiple of 32
    const size_t nt0  = col0 >> 4;
    const size_t base0 = (((size_t)(o >> 5) * NTILES + nt0) * 32 +
                          (size_t)((kp >> 4) * 16)) * 16 + (kp & 15);
    const size_t base1 = base0 + 512;                          // next tile
    float mx = 0.0f;
#pragma unroll
    for (int j = 0; j < 16; ++j)
        mx = fmaxf(mx, fmaxf((float)Y[base0 + (size_t)j * 16] * scale + shift, 0.0f));
#pragma unroll
    for (int j = 0; j < 16; ++j)
        mx = fmaxf(mx, fmaxf((float)Y[base1 + (size_t)j * 16] * scale + shift, 0.0f));
    out[((size_t)b * 128 + o) * NPOINT + s] = mx;
}

// ---------------------------------------------------------------------------
// Host launcher
// ---------------------------------------------------------------------------
extern "C" void kernel_launch(void* const* d_in, const int* in_sizes, int n_in,
                              void* d_out, int out_size, void* d_ws, size_t ws_size,
                              hipStream_t stream) {
    const float* xyz    = (const float*)d_in[0];
    const float* points = (const float*)d_in[1];
    const float* W0 = (const float*)d_in[2];
    const float* g0 = (const float*)d_in[4];
    const float* be0 = (const float*)d_in[5];
    const float* W1 = (const float*)d_in[6];
    const float* g1 = (const float*)d_in[8];
    const float* be1 = (const float*)d_in[9];
    const float* W2 = (const float*)d_in[10];
    const float* g2 = (const float*)d_in[12];
    const float* be2 = (const float*)d_in[13];

    float* out      = (float*)d_out;
    float* new_xyz  = out;
    float* out_feat = out + (size_t)BATCH * NPOINT * 3;

    // workspace: region A (96 rows f16: X0, later Y1) | region B (128 rows
    // f16: Y0, later Y2) | packed A fragments | stats | BN scale/shift
    char* ws = (char*)d_ws;
    const size_t bytesA = (size_t)KPAD0 * NCOLS * sizeof(f16);   // 100.7 MB
    const size_t bytesB = (size_t)128   * NCOLS * sizeof(f16);   // 134.2 MB
    f16* regA = (f16*)ws;
    f16* regB = (f16*)(ws + bytesA);
    char* pk  = ws + bytesA + bytesB;
    f16* A0 = (f16*)pk;                         // 4*3*512 halves
    f16* A1 = (f16*)(pk + 12288);               // 4*2*512 halves
    f16* A2 = (f16*)(pk + 12288 + 8192);        // 8*2*512 halves
    float* stats = (float*)(pk + 36864);
    float* sum0 = stats + 0,   * sq0 = stats + 64;
    float* sum1 = stats + 128, * sq1 = stats + 192;
    float* sum2 = stats + 256, * sq2 = stats + 384;
    float* sc0 = stats + 512, * sh0 = stats + 576;
    float* sc1 = stats + 640, * sh1 = stats + 704;
    float* sc2 = stats + 768, * sh2 = stats + 896;

    zero_stats_kernel<<<1, 512, 0, stream>>>(stats);
    pack_w_kernel<<<24, 256, 0, stream>>>(W0, A0, 4, 3, 67);
    pack_w_kernel<<<16, 256, 0, stream>>>(W1, A1, 4, 2, 64);
    pack_w_kernel<<<32, 256, 0, stream>>>(W2, A2, 8, 2, 64);

    fps_kernel<<<BATCH, 1024, 0, stream>>>(xyz, new_xyz);
    group_kernel<<<(BATCH * NPOINT) / 8, 256, 0, stream>>>(xyz, points, new_xyz, regA);

    const int nquads = (int)(NTILES / 4);                 // 8192
    // layer 0: 64 x 67 (K padded 96): X0(A) -> Y0(B)
    gemm_bnstat_kernel<3, 4><<<(4 * nquads) / 8, 256, 0, stream>>>(A0, regA, regB, sum0, sq0);
    finalize_stats_kernel<<<1, 128, 0, stream>>>(sum0, sq0, g0, be0, sc0, sh0, 64);
    bn_relu_kernel<<<(int)((2 * NTILES * 32) / 256), 256, 0, stream>>>(regB, sc0, sh0);
    // layer 1: 64 x 64: H0(B) -> Y1(A)
    gemm_bnstat_kernel<2, 4><<<(4 * nquads) / 8, 256, 0, stream>>>(A1, regB, regA, sum1, sq1);
    finalize_stats_kernel<<<1, 128, 0, stream>>>(sum1, sq1, g1, be1, sc1, sh1, 64);
    bn_relu_kernel<<<(int)((2 * NTILES * 32) / 256), 256, 0, stream>>>(regA, sc1, sh1);
    // layer 2: 128 x 64: H1(A) -> Y2(B)
    gemm_bnstat_kernel<2, 8><<<(8 * nquads) / 8, 256, 0, stream>>>(A2, regA, regB, sum2, sq2);
    finalize_stats_kernel<<<1, 128, 0, stream>>>(sum2, sq2, g2, be2, sc2, sh2, 128);

    bn_relu_maxpool_kernel<<<(int)(((size_t)BATCH * 128 * NPOINT) / 256), 256, 0, stream>>>(
        regB, sc2, sh2, out_feat);
}
